// TopKSparseAutoencoder_35055523070102
// MI455X (gfx1250) — compile-verified, matrix operand order checked by reading, not run on hardware
//
#include <hip/hip_runtime.h>

typedef _Float16 v16h __attribute__((ext_vector_type(16)));
typedef _Float16 v8h  __attribute__((ext_vector_type(8)));
typedef float    v8f  __attribute__((ext_vector_type(8)));

#define B_SZ   8192
#define D_IN   2048
#define D_HID  16384
#define K_TOP  64

// ---------------- Kernel A: features = relu(x @ W_enc + b_enc) ----------------
// f16 WMMA with f32 accumulation. 128x128 tile, K-step 32, 8 waves (4 M x 2 N),
// each wave = 2x4 fragments of 16x16. Software-pipelined: global loads for
// tile k+1 staged in registers while tile k is consumed from LDS; double-
// buffered LDS -> one barrier per K-step.
#define TM 128
#define TN 128
#define TK 32
#define LDSK 48   // padded K stride (elements); 96B rows keep 16B alignment

__global__ __launch_bounds__(256)
void enc_gemm_wmma(const float* __restrict__ x, const float* __restrict__ W,
                   const float* __restrict__ bias, float* __restrict__ feat) {
    __shared__ _Float16 lA[2][TM * LDSK];   // [buf][m][k]
    __shared__ _Float16 lB[2][TN * LDSK];   // [buf][n][k] (transposed on store)

    const int tid  = threadIdx.x;
    const int lane = tid & 31;
    const int wave = tid >> 5;
    const int wm   = wave >> 1;    // 0..3 : M direction
    const int wn   = wave & 1;     // 0..1 : N direction
    const int half = lane >> 4;    // 0/1  : lane half
    const int l16  = lane & 15;

    const int bm = blockIdx.y * TM;
    const int bn = blockIdx.x * TN;

    v8f acc[2][4];
    #pragma unroll
    for (int mi = 0; mi < 2; ++mi)
        #pragma unroll
        for (int ni = 0; ni < 4; ++ni)
            #pragma unroll
            for (int e = 0; e < 8; ++e) acc[mi][ni][e] = 0.0f;

    float4 ra[4];   // staged A tile (this thread's share)
    float4 rb[4];   // staged B tile

    // Stage tile at k0 into registers (global -> VGPR).
    auto load_tiles = [&](int k0) {
        #pragma unroll
        for (int i = 0; i < 4; ++i) {
            int f   = tid + i * 256;      // float4 id 0..1023
            int row = f >> 3;             // 0..127
            int kq  = f & 7;              // 0..7
            ra[i] = reinterpret_cast<const float4*>(
                        x + (size_t)(bm + row) * D_IN + k0)[kq];
        }
        #pragma unroll
        for (int i = 0; i < 4; ++i) {
            int f  = tid + i * 256;
            int kr = f >> 5;              // 0..31
            int nq = f & 31;              // 0..31 (groups of 4 columns)
            rb[i] = reinterpret_cast<const float4*>(
                        W + (size_t)(k0 + kr) * D_HID + bn)[nq];
        }
    };

    // Convert staged registers to f16 and store into LDS buffer `buf`.
    auto store_tiles = [&](int buf) {
        #pragma unroll
        for (int i = 0; i < 4; ++i) {
            int f   = tid + i * 256;
            int row = f >> 3;
            int kq  = f & 7;
            _Float16* dst = &lA[buf][row * LDSK + kq * 4];
            dst[0] = (_Float16)ra[i].x; dst[1] = (_Float16)ra[i].y;
            dst[2] = (_Float16)ra[i].z; dst[3] = (_Float16)ra[i].w;
        }
        #pragma unroll
        for (int i = 0; i < 4; ++i) {
            int f  = tid + i * 256;
            int kr = f >> 5;
            int nq = f & 31;
            lB[buf][(nq * 4 + 0) * LDSK + kr] = (_Float16)rb[i].x;
            lB[buf][(nq * 4 + 1) * LDSK + kr] = (_Float16)rb[i].y;
            lB[buf][(nq * 4 + 2) * LDSK + kr] = (_Float16)rb[i].z;
            lB[buf][(nq * 4 + 3) * LDSK + kr] = (_Float16)rb[i].w;
        }
    };

    // Prologue: fill buffer 0.
    load_tiles(0);
    store_tiles(0);
    __syncthreads();

    const int NK = D_IN / TK;   // 64
    for (int kk = 0; kk < NK; ++kk) {
        const int cur = kk & 1;

        // Issue next tile's global loads early (overlaps with WMMA below).
        if (kk + 1 < NK) load_tiles((kk + 1) * TK);

        // ---- fragment loads per the 16-bit A/B VGPR layout ----
        // lanes 0-15: K 0..7 in v0..3, K 16..23 in v4..7
        // lanes 16-31: K 8..15 in v0..3, K 24..31 in v4..7
        union Frag { v16h v; v8h h[2]; };
        Frag a[2], b[4];
        #pragma unroll
        for (int mi = 0; mi < 2; ++mi) {
            int r = wm * 32 + mi * 16 + l16;
            a[mi].h[0] = *reinterpret_cast<const v8h*>(&lA[cur][r * LDSK + half * 8]);
            a[mi].h[1] = *reinterpret_cast<const v8h*>(&lA[cur][r * LDSK + 16 + half * 8]);
        }
        #pragma unroll
        for (int ni = 0; ni < 4; ++ni) {
            int c = wn * 64 + ni * 16 + l16;
            b[ni].h[0] = *reinterpret_cast<const v8h*>(&lB[cur][c * LDSK + half * 8]);
            b[ni].h[1] = *reinterpret_cast<const v8h*>(&lB[cur][c * LDSK + 16 + half * 8]);
        }
        #pragma unroll
        for (int mi = 0; mi < 2; ++mi)
            #pragma unroll
            for (int ni = 0; ni < 4; ++ni)
                acc[mi][ni] = __builtin_amdgcn_wmma_f32_16x16x32_f16(
                    false, a[mi].v, false, b[ni].v, (short)0, acc[mi][ni],
                    false, false);

        // Fill the other buffer for the next iteration (no conflict with the
        // buffer read above -> a single end-of-iteration barrier suffices).
        if (kk + 1 < NK) store_tiles(cur ^ 1);
        __syncthreads();
    }

    // ---- epilogue: bias + relu; C layout: VGPR r -> M = r + 8*half ----
    #pragma unroll
    for (int mi = 0; mi < 2; ++mi) {
        #pragma unroll
        for (int ni = 0; ni < 4; ++ni) {
            int col = bn + wn * 64 + ni * 16 + l16;
            float bv = bias[col];
            #pragma unroll
            for (int r = 0; r < 8; ++r) {
                int row = bm + wm * 32 + mi * 16 + half * 8 + r;
                float v = acc[mi][ni][r] + bv;
                feat[(size_t)row * D_HID + col] = v > 0.0f ? v : 0.0f;
            }
        }
    }
}

// ---------------- Kernel B: per-row radix top-K select ----------------
// All values >= +0 after relu, so uint bit ordering == float ordering.
// 4 byte-wise histogram passes find the K-th largest value exactly; final
// sweep zeroes everything below it (ties capped atomically to exactly K)
// and emits a compact (idx, val) list for the decoder.
__global__ __launch_bounds__(256)
void topk_select(float* __restrict__ feat, int* __restrict__ idxL,
                 float* __restrict__ valL) {
    __shared__ unsigned hist[256];
    __shared__ unsigned sh_prefix;
    __shared__ unsigned sh_krem;
    __shared__ int sh_tie;
    __shared__ int sh_out;

    const int row = blockIdx.x;
    const int tid = threadIdx.x;
    float* frow = feat + (size_t)row * D_HID;

    if (tid == 0) { sh_prefix = 0u; sh_krem = K_TOP; sh_tie = 0; sh_out = 0; }
    __syncthreads();

    unsigned mask = 0u;
    for (int shift = 24; shift >= 0; shift -= 8) {
        hist[tid] = 0u;
        __syncthreads();
        unsigned pfx = sh_prefix;
        for (int i = tid; i < D_HID; i += 256) {
            unsigned u = __float_as_uint(frow[i]);
            if ((u & mask) == pfx)
                atomicAdd(&hist[(u >> shift) & 255u], 1u);
        }
        __syncthreads();
        if (tid == 0) {
            unsigned krem = sh_krem, cum = 0u; int chosen = 0;
            for (int bkt = 255; bkt >= 0; --bkt) {
                unsigned c = hist[bkt];
                if (cum + c >= krem) { chosen = bkt; break; }
                cum += c;
            }
            sh_prefix |= ((unsigned)chosen) << shift;
            sh_krem = krem - cum;      // ties still needed inside chosen bucket
        }
        __syncthreads();
        mask |= 0xFFu << shift;
    }

    const unsigned thr  = sh_prefix;   // exact bits of K-th largest value
    const unsigned krem = sh_krem;     // how many threshold-equal elems to keep

    for (int i = tid; i < D_HID; i += 256) {
        float v = frow[i];
        unsigned u = __float_as_uint(v);
        bool keep = false;
        if (u > thr) {
            keep = true;
        } else if (u == thr) {
            int t = atomicAdd(&sh_tie, 1);
            if (t < (int)krem) keep = true;
        }
        if (keep) {
            int s = atomicAdd(&sh_out, 1);
            idxL[(size_t)row * K_TOP + s] = i;
            valL[(size_t)row * K_TOP + s] = v;
        } else {
            frow[i] = 0.0f;
        }
    }
}

// ---------------- Kernel C: recon[b,:] = sum_t val[b,t] * W_dec[idx[b,t],:] --
// One block per row; W_dec (128 MB) stays L2-resident (192 MB L2).
__global__ __launch_bounds__(256)
void decode_scatter(const float* __restrict__ Wd, const int* __restrict__ idxL,
                    const float* __restrict__ valL, float* __restrict__ recon) {
    __shared__ int   sidx[K_TOP];
    __shared__ float sval[K_TOP];
    const int row = blockIdx.x;
    const int tid = threadIdx.x;
    if (tid < K_TOP) {
        sidx[tid] = idxL[(size_t)row * K_TOP + tid];
        sval[tid] = valL[(size_t)row * K_TOP + tid];
    }
    __syncthreads();

    float4 a0 = {0.f, 0.f, 0.f, 0.f};
    float4 a1 = {0.f, 0.f, 0.f, 0.f};
    const int c0 = tid * 8;   // 2048 cols / 256 threads = 8 per thread
    #pragma unroll 4
    for (int t = 0; t < K_TOP; ++t) {
        float v = sval[t];
        const float4* wr =
            reinterpret_cast<const float4*>(Wd + (size_t)sidx[t] * D_IN + c0);
        float4 w0 = wr[0], w1 = wr[1];
        a0.x += v * w0.x; a0.y += v * w0.y; a0.z += v * w0.z; a0.w += v * w0.w;
        a1.x += v * w1.x; a1.y += v * w1.y; a1.z += v * w1.z; a1.w += v * w1.w;
    }
    float4* out = reinterpret_cast<float4*>(recon + (size_t)row * D_IN + c0);
    out[0] = a0; out[1] = a1;
}

extern "C" void kernel_launch(void* const* d_in, const int* in_sizes, int n_in,
                              void* d_out, int out_size, void* d_ws, size_t ws_size,
                              hipStream_t stream) {
    (void)in_sizes; (void)n_in; (void)out_size; (void)ws_size;
    const float* x     = (const float*)d_in[0];
    const float* W_enc = (const float*)d_in[1];
    const float* b_enc = (const float*)d_in[2];
    const float* W_dec = (const float*)d_in[3];
    // d_in[4] is k (scalar == 64), hardcoded as K_TOP.

    float* recon  = (float*)d_out;                       // [B, D_IN]
    float* sparse = recon + (size_t)B_SZ * D_IN;         // [B, D_HID]

    int*   idxL = (int*)d_ws;                                        // [B, K]
    float* valL = (float*)((char*)d_ws + (size_t)B_SZ * K_TOP * 4);  // [B, K]

    dim3 gA(D_HID / TN, B_SZ / TM);   // 128 x 64 blocks
    enc_gemm_wmma<<<gA, 256, 0, stream>>>(x, W_enc, b_enc, sparse);
    topk_select<<<B_SZ, 256, 0, stream>>>(sparse, idxL, valL);
    decode_scatter<<<B_SZ, 256, 0, stream>>>(W_dec, idxL, valL, recon);
}
